// ProgressiveFocusedSparseAttention_87883620810890
// MI455X (gfx1250) — matrix operand order repaired and checked by reading, not compiled, not run on hardware
//
#include <hip/hip_runtime.h>
#include <hip/hip_bf16.h>

// ---------------- problem constants (fixed by the reference) ----------------
#define B_    4
#define N_    4096
#define DIM_  512
#define H_    8
#define HD_   64
#define GS_   128
#define WS_   256
#define NG_   32
#define NLBL  8

typedef __attribute__((ext_vector_type(16))) __bf16 v16bf;
typedef __attribute__((ext_vector_type(8)))  __bf16 v8bf;
typedef __attribute__((ext_vector_type(8)))  float  v8f;
typedef __attribute__((ext_vector_type(4)))  unsigned int u32x4;
typedef __attribute__((ext_vector_type(8)))  int i32x8;
typedef __attribute__((ext_vector_type(4)))  int i32x4;
typedef __attribute__((ext_vector_type(4)))  int v4i;

// ---------------- CDNA5 feature probes (fallbacks keep compile green) -------
#if defined(__has_builtin)
#if __has_builtin(__builtin_amdgcn_global_load_async_to_lds_b128)
#define HAVE_ASYNC_LDS 1
#endif
#if __has_builtin(__builtin_amdgcn_s_wait_asynccnt)
#define HAVE_WAIT_ASYNC 1
#endif
#if __has_builtin(__builtin_amdgcn_tensor_load_to_lds)
#define HAVE_TDM 1
#endif
#endif

__device__ inline unsigned lds_addr_of(const void* p) {
  return (unsigned)(unsigned long long)(__attribute__((address_space(3))) const void*)p;
}

// 16-byte global -> LDS copy (async-to-LDS when available)
__device__ inline void g2l_16(__bf16* dst, const __bf16* src) {
#if HAVE_ASYNC_LDS
  __builtin_amdgcn_global_load_async_to_lds_b128(
      (__attribute__((address_space(1))) v4i*)(void*)(__bf16*)src,
      (__attribute__((address_space(3))) v4i*)(void*)dst, 0, 0);
#else
  *(uint4*)dst = *(const uint4*)src;
#endif
}
__device__ inline void wait_async_lds() {
#if HAVE_ASYNC_LDS
#if HAVE_WAIT_ASYNC
  __builtin_amdgcn_s_wait_asynccnt(0);
#else
  asm volatile("s_wait_asynccnt 0x0" ::: "memory");
#endif
#endif
}

// key-window row index: window = [group g | group g+1], last group mirrors tail
__device__ inline int win_row(int g, int w) {
  if (w < GS_) return g * GS_ + w;
  int j = w - GS_;
  return (g < NG_ - 1) ? ((g + 1) * GS_ + j) : (N_ - 1 - j);
}

// ---------------- elementwise f32 -> bf16 (row-major, for activations) ------
__global__ void cvt_bf16_kernel(const float* __restrict__ in,
                                __bf16* __restrict__ out, size_t n) {
  size_t i = (size_t)blockIdx.x * blockDim.x + threadIdx.x;
  if (i < n) out[i] = (__bf16)in[i];
}

// ----- f32 -> bf16 weight conversion into WMMA B-fragment order -------------
// out[(k/32)*(N/16) + n/16][lane][e]  with lane = (n&15) + 16*((k&31)>>4),
// e = k&15  (contiguous k = 16*half + e order, same as the A fragments).
__global__ void cvt_swz_b_kernel(const float* __restrict__ W,
                                 __bf16* __restrict__ out, int K, int N) {
  int i = blockIdx.x * blockDim.x + threadIdx.x;
  if (i >= K * N) return;
  int k = i / N, n = i % N;
  int sub = (k >> 5) * (N >> 4) + (n >> 4);
  int lane = (n & 15) + 16 * ((k & 31) >> 4);
  int e = k & 15;
  out[(size_t)sub * 512 + lane * 16 + e] = (__bf16)W[i];
}

// ---------------- LDS-free direct-fragment bf16 GEMM ------------------------
// A[M,K] row-major bf16, Bsw pre-swizzled fragment order, C[M,N].
// 128 thr = 4 waves, each wave 32x32, block tile 64x64. K multiple of 32.
template <bool OUT_BF16>
__global__ void gemm_bf16_direct(const __bf16* __restrict__ A,
                                 const __bf16* __restrict__ Bsw,
                                 void* __restrict__ C, int M, int N, int K) {
  int tid = threadIdx.x, lane = tid & 31, wave = tid >> 5;
  int m0 = blockIdx.x * 64 + (wave >> 1) * 32;
  int n0 = blockIdx.y * 64 + (wave & 1) * 32;
  int r = lane & 15, hi = lane >> 4;
  v8f acc[2][2] = {};
  for (int k0 = 0; k0 < K; k0 += 32) {
    v16bf af[2], bfr[2];
#pragma unroll
    for (int i = 0; i < 2; ++i) {
      const __bf16* base = A + (size_t)(m0 + 16 * i + r) * K + k0 + hi * 16;
      v8bf lo = *(const v8bf*)base;
      v8bf hh = *(const v8bf*)(base + 8);
      af[i] = __builtin_shufflevector(lo, hh, 0, 1, 2, 3, 4, 5, 6, 7,
                                      8, 9, 10, 11, 12, 13, 14, 15);
    }
#pragma unroll
    for (int j = 0; j < 2; ++j) {
      const __bf16* bb =
          Bsw + ((size_t)(k0 >> 5) * (N >> 4) + (n0 >> 4) + j) * 512 + lane * 16;
      bfr[j] = *(const v16bf*)bb;
    }
#pragma unroll
    for (int i = 0; i < 2; ++i)
#pragma unroll
      for (int j = 0; j < 2; ++j)
        acc[i][j] = __builtin_amdgcn_wmma_f32_16x16x32_bf16(
            false, af[i], false, bfr[j], (short)0, acc[i][j], false, false);
  }
  int n = lane & 15;
#pragma unroll
  for (int i = 0; i < 2; ++i)
#pragma unroll
    for (int j = 0; j < 2; ++j)
#pragma unroll
      for (int rr = 0; rr < 8; ++rr) {
        size_t idx = (size_t)(m0 + 16 * i + hi * 8 + rr) * N + n0 + 16 * j + n;
        if constexpr (OUT_BF16) ((__bf16*)C)[idx] = (__bf16)acc[i][j][rr];
        else                    ((float*)C)[idx]  = acc[i][j][rr];
      }
}

// ---------------- per-group focus statistics --------------------------------
__global__ void group_stats_kernel(const int* __restrict__ labels,
                                   const float* __restrict__ scores,
                                   int* __restrict__ same_keep,
                                   int* __restrict__ cross_keep) {
  int idx = blockIdx.x * blockDim.x + threadIdx.x;
  if (idx >= B_ * NG_) return;
  int b = idx / NG_, g = idx % NG_;
  const int* lp = labels + b * N_ + g * GS_;
  const float* sp = scores + b * N_ + g * GS_;
  int cnt[NLBL];
#pragma unroll
  for (int i = 0; i < NLBL; ++i) cnt[i] = 0;
  float sum = 0.f, sumsq = 0.f;
  for (int i = 0; i < GS_; ++i) {
    int l = lp[i];
    if (l >= 0 && l < NLBL) cnt[l]++;
    float s = sp[i];
    sum += s; sumsq += s * s;
  }
  int mode = 0;
#pragma unroll
  for (int l = 1; l < NLBL; ++l) if (cnt[l] > cnt[mode]) mode = l;
  float purity = (float)cnt[mode] * (1.0f / GS_);
  float mean = sum * (1.0f / GS_);
  float var = sumsq * (1.0f / GS_) - mean * mean;
  float focus = fminf(fmaxf(0.5f + 0.25f * purity - 0.25f * var, 0.25f), 0.75f);
  int keep = (int)ceilf(focus * (float)WS_);
  keep = min(max(keep, 1), WS_);
  int ck = (int)rintf((float)keep * 0.125f);
  ck = (keep > 1) ? max(ck, 1) : 0;
  same_keep[idx] = max(keep - ck, 1);
  cross_keep[idx] = ck;
}

// ---------------- logits: per (b,g,h,half) 128x128x64 WMMA GEMM -------------
// Qs is stored in A-fragment order [kstep][sub16][lane][16]; Ks row-major
// (128B rows, 32B aligned fragment slices -> wide ds_load_b128).
__global__ void logits_kernel(const __bf16* __restrict__ qb,
                              const __bf16* __restrict__ kb,
                              float* __restrict__ logits) {
  __shared__ __bf16 Qs[2][8][32][16];  // 16 KB
  __shared__ __bf16 Ks[128][64];       // 16 KB
  int bid = blockIdx.x;
  int half = bid & 1;
  int tile = bid >> 1;                       // [b][g][h]
  int h = tile % H_, g = (tile / H_) % NG_, b = tile / (H_ * NG_);
  int tid = threadIdx.x, lane = tid & 31, wave = tid >> 5;
  const __bf16* qbase = qb + (size_t)b * N_ * DIM_ + h * HD_;
  const __bf16* kbase = kb + (size_t)b * N_ * DIM_ + h * HD_;

  // ---- stage K tile (128 rows x 64 dims) ----
  bool contig = (half == 0) || (g < NG_ - 1);
  bool tdm_done = false;
#if HAVE_TDM
  if (contig) {
    if (wave == 0) {
      int row0 = (half == 0) ? g * GS_ : (g + 1) * GS_;
      unsigned long long ga =
          (unsigned long long)(const void*)(kbase + (size_t)row0 * DIM_);
      u32x4 g0 = {1u,                                  // count=1
                  lds_addr_of(&Ks[0][0]),              // lds_addr
                  (unsigned)(ga & 0xffffffffu),
                  (unsigned)((ga >> 32) & 0x01ffffffu) | 0x80000000u};  // type=2
      i32x8 g1 = {(int)(1u << 16),        // data_size = 2 bytes
                  (int)(64u << 16),       // tensor_dim0 = 64
                  (int)(128u << 16),      // tensor_dim1 = 128
                  (int)(64u << 16),       // tile_dim0 = 64
                  (int)128,               // tile_dim1 = 128
                  (int)512,               // tensor_dim0_stride = DIM_
                  0, 0};
      i32x4 gz4 = {0, 0, 0, 0};
      i32x8 gz8 = {0, 0, 0, 0, 0, 0, 0, 0};
      __builtin_amdgcn_tensor_load_to_lds(g0, g1, gz4, gz4, gz8, 0);
      __builtin_amdgcn_s_wait_tensorcnt(0);
    }
    tdm_done = true;
  }
#endif
  if (!tdm_done) {
    for (int idx = tid; idx < 128 * 8; idx += 256) {
      int key = idx >> 3, oct = idx & 7;
      int row = win_row(g, half * GS_ + key);
      g2l_16(&Ks[key][oct * 8], kbase + (size_t)row * DIM_ + oct * 8);
    }
  }
  // ---- stage Q tile into fragment order (16B contiguous both sides) ----
  for (int idx = tid; idx < 128 * 8; idx += 256) {
    int r = idx >> 3, o = idx & 7;
    int kk = o >> 2, oo = o & 3;
    int ln = (r & 15) + 16 * (oo >> 1);
    int eb = (oo & 1) ? 8 : 0;
    g2l_16(&Qs[kk][r >> 4][ln][eb], qbase + (size_t)(g * GS_ + r) * DIM_ + o * 8);
  }
  wait_async_lds();
  __syncthreads();

  int wr = (wave >> 1) * 32;  // 4 row groups of 32
  int wc = (wave & 1) * 64;   // 2 col groups of 64
  int n = lane & 15, hi = lane >> 4;
  v8f acc[2][4] = {};
#pragma unroll
  for (int kk = 0; kk < 2; ++kk) {
    v16bf af[2], bfr[4];
#pragma unroll
    for (int i = 0; i < 2; ++i)
      af[i] = *(const v16bf*)&Qs[kk][(wr >> 4) + i][lane][0];
#pragma unroll
    for (int j = 0; j < 4; ++j)
      bfr[j] = *(const v16bf*)&Ks[wc + 16 * j + n][kk * 32 + hi * 16];
#pragma unroll
    for (int i = 0; i < 2; ++i)
#pragma unroll
      for (int j = 0; j < 4; ++j)
        acc[i][j] = __builtin_amdgcn_wmma_f32_16x16x32_bf16(
            false, af[i], false, bfr[j], (short)0, acc[i][j], false, false);
  }
  float* out = logits + (size_t)tile * GS_ * WS_;
#pragma unroll
  for (int i = 0; i < 2; ++i)
#pragma unroll
    for (int j = 0; j < 4; ++j)
#pragma unroll
      for (int rr = 0; rr < 8; ++rr)
        out[(wr + 16 * i + hi * 8 + rr) * WS_ + half * GS_ + wc + 16 * j + n] =
            acc[i][j][rr] * 0.125f;  // 1/sqrt(64)
}

// ---------------- block reductions (256 threads) ----------------------------
__device__ inline float bred_max(float v, float* red, int t) {
  red[t] = v; __syncthreads();
  for (int s = 128; s > 0; s >>= 1) {
    if (t < s) red[t] = fmaxf(red[t], red[t + s]);
    __syncthreads();
  }
  float r = red[0]; __syncthreads();
  return r;
}
__device__ inline float bred_sum(float v, float* red, int t) {
  red[t] = v; __syncthreads();
  for (int s = 128; s > 0; s >>= 1) {
    if (t < s) red[t] = red[t] + red[t + s];
    __syncthreads();
  }
  float r = red[0]; __syncthreads();
  return r;
}

// ---------------- softmax + rank-based top-k mask + sparse re-softmax -------
__global__ void mask_softmax_kernel(const float* __restrict__ logits,
                                    const int* __restrict__ labels,
                                    const float* __restrict__ scores,
                                    const int* __restrict__ same_keep,
                                    const int* __restrict__ cross_keep,
                                    float* __restrict__ attn_out) {
  __shared__ int   kl[WS_];
  __shared__ float ksc[WS_];
  __shared__ int   ql[GS_];
  __shared__ float qsc[GS_];
  __shared__ float sv[WS_], cv[WS_], red[WS_];
  int bid = blockIdx.x;
  int g = (bid / H_) % NG_, b = bid / (H_ * NG_);
  int t = threadIdx.x;  // 256
  if (t < GS_) {
    ql[t]  = labels[b * N_ + g * GS_ + t];
    qsc[t] = scores[b * N_ + g * GS_ + t];
  }
  {
    int row = win_row(g, t);
    kl[t]  = labels[b * N_ + row];
    ksc[t] = scores[b * N_ + row];
  }
  __syncthreads();
  int sk = same_keep[b * NG_ + g], ck = cross_keep[b * NG_ + g];
  const float* lrow = logits + (size_t)bid * GS_ * WS_;
  float* orow = attn_out + (size_t)bid * GS_ * WS_;
  for (int r = 0; r < GS_; ++r) {
    float l = lrow[r * WS_ + t];
    float mx = bred_max(l, red, t);
    float e = __expf(l - mx);
    float s = bred_sum(e, red, t);
    float ah = e / s;
    bool same = (ql[r] == kl[t]);
    float sval = same ? ah : -10000.0f;
    float cval = ah * qsc[r] * ksc[t];
    sv[t] = sval; cv[t] = cval;
    __syncthreads();
    int rs = 0, rc = 0;
    for (int j = 0; j < WS_; ++j) {
      float a = sv[j];
      rs += ((a > sval) || (a == sval && j < t)) ? 1 : 0;
      float c = cv[j];
      rc += ((c > cval) || (c == cval && j < t)) ? 1 : 0;
    }
    bool m = (same && rs < sk) || (!same && rc < ck);
    float sl = m ? l : -10000.0f;
    float mx2 = bred_max(sl, red, t);
    float e2 = __expf(sl - mx2);
    float s2 = bred_sum(e2, red, t);
    float a2 = e2 / s2;
    a2 = m ? a2 : 0.0f;
    float s3 = bred_sum(a2, red, t);
    a2 = a2 / (s3 + 1e-9f);
    orow[r * WS_ + t] = a2;
    __syncthreads();
  }
}

// ---------------- attn_local @ V : per (b,g,h) 128x64x256 WMMA GEMM ---------
// A tile converted f32->bf16 into fragment order; V staged transposed so the
// B fragment is a contiguous 32B per-lane LDS load.
__global__ void attn_v_kernel(const float* __restrict__ attn,
                              const __bf16* __restrict__ vb,
                              __bf16* __restrict__ out_pre) {
  __shared__ __bf16 As[8][32][16];  // 128x32 in fragment order (8 KB)
  __shared__ __bf16 Vt[64][32];     // V^T tile (4 KB)
  int bid = blockIdx.x;
  int h = bid % H_, g = (bid / H_) % NG_, b = bid / (H_ * NG_);
  int tid = threadIdx.x, lane = tid & 31, wave = tid >> 5;  // 8 waves x 16 rows
  const float* abase = attn + (size_t)bid * GS_ * WS_;
  const __bf16* vbase = vb + (size_t)b * N_ * DIM_ + h * HD_;
  int n = lane & 15, hi = lane >> 4;
  v8f acc[4] = {};
  for (int k0 = 0; k0 < WS_; k0 += 32) {
    for (int idx = tid; idx < 128 * 4; idx += 256) {
      int r = idx >> 2, oo = idx & 3;
      const float4* s = (const float4*)(abase + r * WS_ + k0 + oo * 8);
      float4 x0 = s[0], x1 = s[1];
      __bf16 tmp[8] = {(__bf16)x0.x, (__bf16)x0.y, (__bf16)x0.z, (__bf16)x0.w,
                       (__bf16)x1.x, (__bf16)x1.y, (__bf16)x1.z, (__bf16)x1.w};
      int ln = (r & 15) + 16 * (oo >> 1);
      int eb = (oo & 1) ? 8 : 0;
      *(uint4*)&As[r >> 4][ln][eb] = *(uint4*)tmp;
    }
    for (int idx = tid; idx < 32 * 8; idx += 256) {
      int k = idx >> 3, dd = (idx & 7) * 8;
      const __bf16* src = vbase + (size_t)win_row(g, k0 + k) * DIM_ + dd;
#pragma unroll
      for (int tt = 0; tt < 8; ++tt) Vt[dd + tt][k] = src[tt];
    }
    __syncthreads();
    v16bf a = *(const v16bf*)&As[wave][lane][0];
    v16bf bfr[4];
#pragma unroll
    for (int j = 0; j < 4; ++j)
      bfr[j] = *(const v16bf*)&Vt[16 * j + n][hi * 16];
#pragma unroll
    for (int j = 0; j < 4; ++j)
      acc[j] = __builtin_amdgcn_wmma_f32_16x16x32_bf16(
          false, a, false, bfr[j], (short)0, acc[j], false, false);
    __syncthreads();
  }
#pragma unroll
  for (int j = 0; j < 4; ++j)
#pragma unroll
    for (int rr = 0; rr < 8; ++rr) {
      int row = g * GS_ + wave * 16 + hi * 8 + rr;
      out_pre[((size_t)b * N_ + row) * DIM_ + h * HD_ + 16 * j + n] =
          (__bf16)acc[j][rr];
    }
}

// ---------------- host orchestration ----------------------------------------
extern "C" void kernel_launch(void* const* d_in, const int* in_sizes, int n_in,
                              void* d_out, int out_size, void* d_ws, size_t ws_size,
                              hipStream_t stream) {
  const float* x      = (const float*)d_in[0];
  const int*   labels = (const int*)d_in[1];
  const float* scores = (const float*)d_in[2];
  const float* Wq     = (const float*)d_in[3];
  const float* Wk     = (const float*)d_in[4];
  const float* Wv     = (const float*)d_in[5];
  const float* Wp     = (const float*)d_in[6];

  const size_t XN = (size_t)B_ * N_ * DIM_;             // 8,388,608
  const size_t WN = (size_t)DIM_ * DIM_;                // 262,144
  const size_t LN = (size_t)B_ * NG_ * H_ * GS_ * WS_;  // 33,554,432

  char* p = (char*)d_ws;
  auto alloc = [&](size_t bytes) {
    void* r = (void*)p;
    p += (bytes + 255) & ~(size_t)255;
    return r;
  };
  __bf16* xb   = (__bf16*)alloc(XN * 2);
  __bf16* wqb  = (__bf16*)alloc(WN * 2);
  __bf16* wkb  = (__bf16*)alloc(WN * 2);
  __bf16* wvb  = (__bf16*)alloc(WN * 2);
  __bf16* wpb  = (__bf16*)alloc(WN * 2);
  __bf16* qbuf = (__bf16*)alloc(XN * 2);
  __bf16* kbuf = (__bf16*)alloc(XN * 2);
  __bf16* vbuf = (__bf16*)alloc(XN * 2);
  __bf16* opre = (__bf16*)alloc(XN * 2);
  float*  lgts = (float*)alloc(LN * 4);
  int*    skeep = (int*)alloc(B_ * NG_ * 4);
  int*    ckeep = (int*)alloc(B_ * NG_ * 4);

  // 1) downcast x; swizzle weights into WMMA fragment order
  cvt_bf16_kernel<<<(unsigned)((XN + 255) / 256), 256, 0, stream>>>(x, xb, XN);
  unsigned wg = (unsigned)((WN + 255) / 256);
  cvt_swz_b_kernel<<<wg, 256, 0, stream>>>(Wq, wqb, DIM_, DIM_);
  cvt_swz_b_kernel<<<wg, 256, 0, stream>>>(Wk, wkb, DIM_, DIM_);
  cvt_swz_b_kernel<<<wg, 256, 0, stream>>>(Wv, wvb, DIM_, DIM_);
  cvt_swz_b_kernel<<<wg, 256, 0, stream>>>(Wp, wpb, DIM_, DIM_);

  // 2) Q/K/V projections: LDS-free direct-fragment WMMA GEMMs
  dim3 gp((B_ * N_) / 64, DIM_ / 64);
  gemm_bf16_direct<true><<<gp, 128, 0, stream>>>(xb, wqb, qbuf, B_ * N_, DIM_, DIM_);
  gemm_bf16_direct<true><<<gp, 128, 0, stream>>>(xb, wkb, kbuf, B_ * N_, DIM_, DIM_);
  gemm_bf16_direct<true><<<gp, 128, 0, stream>>>(xb, wvb, vbuf, B_ * N_, DIM_, DIM_);

  // 3) per-group focus ratio -> keep counts
  group_stats_kernel<<<1, 128, 0, stream>>>(labels, scores, skeep, ckeep);

  // 4) windowed QK^T logits (scaled) -> f32 scratch (TDM/async staging)
  logits_kernel<<<B_ * NG_ * H_ * 2, 256, 0, stream>>>(qbuf, kbuf, lgts);

  // 5) softmax + top-k mask + sparse re-softmax -> attn_local (2nd output)
  float* out0 = (float*)d_out;
  float* attn_out = out0 + XN;
  mask_softmax_kernel<<<B_ * NG_ * H_, 256, 0, stream>>>(
      lgts, labels, scores, skeep, ckeep, attn_out);

  // 6) attn_local @ V (windowed) -> pre-projection activations (bf16)
  attn_v_kernel<<<B_ * NG_ * H_, 256, 0, stream>>>(attn_out, vbuf, opre);

  // 7) final projection -> 1st output (f32)
  gemm_bf16_direct<false><<<gp, 128, 0, stream>>>(opre, wpb, out0, B_ * N_, DIM_, DIM_);

  (void)in_sizes; (void)n_in; (void)out_size; (void)ws_size;
}
